// MixtureOfExperts_39943195853562
// MI455X (gfx1250) — matrix operand order repaired and checked by reading, not compiled, not run on hardware
//
#include <hip/hip_runtime.h>
#include <hip/hip_bf16.h>

#define N_TOK 16384
#define DIM   1024
#define NEXP  8

#define BM 64
#define BN 128
#define BK 32
#define NKS (DIM / BK)            // 32 k-steps
#define NTILES (NEXP * NKS)       // 256 weight tiles in the pipelined stream
#define XSTRIDE 2064              // 1024*bf16 = 2048B + 16B pad (rotates banks by 4 dwords/row)
#define WSTRIDE 80                // 32*bf16 = 64B + 16B pad (conflict-free b128)

typedef __bf16 bf16_t;
typedef __bf16 v16bf __attribute__((ext_vector_type(16)));
typedef float  v8f   __attribute__((ext_vector_type(8)));
typedef int    v4i   __attribute__((ext_vector_type(4)));

union Frag { v16bf v; v4i q[2]; };

// ---- CDNA5 async global->LDS path (ASYNCcnt), with safe fallback ----------------
#ifndef __HIP_DEVICE_COMPILE__
#define HAVE_ASYNC 0
#elif __has_builtin(__builtin_amdgcn_global_load_async_to_lds_b128)
#define HAVE_ASYNC 1
#else
#define HAVE_ASYNC 0
#endif

#if HAVE_ASYNC
#if __has_builtin(__builtin_amdgcn_s_wait_asynccnt)
#define WAIT_ASYNC(n) __builtin_amdgcn_s_wait_asynccnt(n)
#else
#define WAIT_ASYNC(n) asm volatile("s_wait_asynccnt %0" ::"n"(n))
#endif
#else
#define WAIT_ASYNC(n) ((void)0)
#endif

#if HAVE_ASYNC
typedef __attribute__((address_space(1))) v4i gv4i_t;   // global 16B vector
typedef __attribute__((address_space(3))) v4i lv4i_t;   // LDS 16B vector
#endif

__device__ __forceinline__ void async_copy16(void* lds_dst, const void* gsrc) {
#if HAVE_ASYNC
  __builtin_amdgcn_global_load_async_to_lds_b128(
      (gv4i_t*)gsrc, (lv4i_t*)lds_dst, 0, 0);
#else
  *(v4i*)lds_dst = *(const v4i*)gsrc;
#endif
}

__device__ __forceinline__ unsigned short f2bf_rne(float f) {
  unsigned int u = __builtin_bit_cast(unsigned int, f);
  u += 0x7FFFu + ((u >> 16) & 1u);
  return (unsigned short)(u >> 16);
}

// ---------------- fp32 -> bf16 conversion (bandwidth-bound pre-pass) ----------------
__global__ __launch_bounds__(256) void cvt_bf16_kernel(const float* __restrict__ in,
                                                       uint2* __restrict__ out, int n4) {
  int i = blockIdx.x * 256 + threadIdx.x;
  if (i >= n4) return;
  float4 f = ((const float4*)in)[i];
  uint2 o;
  o.x = (unsigned)f2bf_rne(f.x) | ((unsigned)f2bf_rne(f.y) << 16);
  o.y = (unsigned)f2bf_rne(f.z) | ((unsigned)f2bf_rne(f.w) << 16);
  out[i] = o;
}

// ---------------- router: logits -> top-2 -> renormalized gates -------------------
// softmax(top2)/sum(top2) == 2-way softmax over the two top logits (denominator cancels)
__global__ __launch_bounds__(256) void router_kernel(const float* __restrict__ x,
                                                     const float* __restrict__ rw,
                                                     const float* __restrict__ rb,
                                                     float* __restrict__ gate) {
  __shared__ float s_w[NEXP * DIM];   // 32 KB, broadcast reads
  for (int i = threadIdx.x; i < NEXP * DIM; i += 256) s_w[i] = rw[i];
  __syncthreads();

  int n = blockIdx.x * 256 + threadIdx.x;
  const float4* xr = (const float4*)(x + (size_t)n * DIM);
  float acc[NEXP];
#pragma unroll
  for (int e = 0; e < NEXP; ++e) acc[e] = 0.f;

  for (int d4 = 0; d4 < DIM / 4; ++d4) {
    float4 xv = xr[d4];
#pragma unroll
    for (int e = 0; e < NEXP; ++e) {
      float4 wv = *(const float4*)&s_w[e * DIM + d4 * 4];
      acc[e] += xv.x * wv.x + xv.y * wv.y + xv.z * wv.z + xv.w * wv.w;
    }
  }
#pragma unroll
  for (int e = 0; e < NEXP; ++e) acc[e] += rb[e];

  int i0 = 0; float l0 = acc[0];
#pragma unroll
  for (int e = 1; e < NEXP; ++e) if (acc[e] > l0) { l0 = acc[e]; i0 = e; }
  int i1 = -1; float l1 = -3.4e38f;
#pragma unroll
  for (int e = 0; e < NEXP; ++e) if (e != i0 && acc[e] > l1) { l1 = acc[e]; i1 = e; }

  float w0 = 1.f / (1.f + expf(l1 - l0));
  float w1 = 1.f - w0;
  float* g = gate + (size_t)n * NEXP;
#pragma unroll
  for (int e = 0; e < NEXP; ++e) g[e] = (e == i0) ? w0 : ((e == i1) ? w1 : 0.f);
}

// ---------------- main MoE GEMM: persistent-x LDS slab + async double-buffered w ----
__global__ __launch_bounds__(256) void moe_gemm_kernel(const bf16_t* __restrict__ xbf,
                                                       const bf16_t* __restrict__ wbf,
                                                       const float* __restrict__ eb,
                                                       const float* __restrict__ gate,
                                                       float* __restrict__ out) {
  __shared__ __align__(16) unsigned char lx[BM * XSTRIDE];       // 129 KB: x slab [64 rows][1024 bf16 + pad]
  __shared__ __align__(16) unsigned char lw[2 * BN * WSTRIDE];   // 20 KB: double-buffered w tile
  __shared__ float lg[NEXP * BM];                                // 2 KB: gates [e][row]

  const int m0 = blockIdx.y * BM;
  const int n0 = blockIdx.x * BN;
  const int tid  = threadIdx.x;
  const int wave = tid >> 5, lane = tid & 31;
  const int wm = wave >> 2, wn = wave & 3;   // 2 x 4 wave grid -> wave tile 32(M) x 32(N)
  const int half = lane >> 4, lm = lane & 15;

  // ---- prologue: async-load whole x slab (64 x 1024 bf16 = 128 KB, 32 chunks/thread)
#pragma unroll 4
  for (int c = tid; c < BM * 128; c += 256) {        // 16B chunks: row = c>>7, pos = c&127
    int row = c >> 7, pos = c & 127;
    async_copy16(lx + row * XSTRIDE + pos * 16,
                 (const char*)(xbf + (size_t)(m0 + row) * DIM) + pos * 16);
  }
  // gates -> LDS [e][row]
  for (int i = tid; i < NEXP * BM; i += 256) {
    int e = i >> 6, row = i & (BM - 1);
    lg[e * BM + row] = gate[(size_t)(m0 + row) * NEXP + e];
  }
  // issue first two weight tiles of the 256-tile (expert x kstep) stream
#pragma unroll
  for (int g = 0; g < 2; ++g) {
    const bf16_t* src = wbf + (size_t)(g >> 5) * DIM * DIM + (size_t)(g & 31) * BK;
    unsigned char* dst = lw + (g & 1) * (BN * WSTRIDE);
#pragma unroll
    for (int c = tid; c < 512; c += 256) {
      int row = c >> 2, pos = c & 3;
      async_copy16(dst + row * WSTRIDE + pos * 16,
                   (const char*)(src + (size_t)(n0 + row) * DIM) + pos * 16);
    }
  }

  v8f fa[4];
#pragma unroll
  for (int t = 0; t < 4; ++t)
#pragma unroll
    for (int r = 0; r < 8; ++r) fa[t][r] = 0.f;

  const unsigned char* basex = lx + (wm * 32 + lm) * XSTRIDE;

  for (int e = 0; e < NEXP; ++e) {
    v8f acc[4];
#pragma unroll
    for (int t = 0; t < 4; ++t)
#pragma unroll
      for (int r = 0; r < 8; ++r) acc[t][r] = 0.f;

#pragma unroll 2
    for (int ks = 0; ks < NKS; ++ks) {
      const int g = e * NKS + ks;                    // linear tile index
      // older of the two in-flight tiles (and the x slab, in-order) has landed
      if (g == NTILES - 1) { WAIT_ASYNC(0); } else { WAIT_ASYNC(2); }
      __syncthreads();

      const unsigned char* lwbuf = lw + (g & 1) * (BN * WSTRIDE);
      Frag a[2], b[2];
#pragma unroll
      for (int mt = 0; mt < 2; ++mt) {
        const unsigned char* p = basex + mt * 16 * XSTRIDE + ks * 64;
        a[mt].q[0] = *(const v4i*)(p + half * 16);        // K = half*8 .. +7
        a[mt].q[1] = *(const v4i*)(p + 32 + half * 16);   // K = 16+half*8 .. +7
      }
#pragma unroll
      for (int nt = 0; nt < 2; ++nt) {
        const unsigned char* p = lwbuf + (wn * 32 + nt * 16 + lm) * WSTRIDE;
        b[nt].q[0] = *(const v4i*)(p + half * 16);        // K = half*8 .. +7
        b[nt].q[1] = *(const v4i*)(p + 32 + half * 16);   // K = 16+half*8 .. +7
      }

#pragma unroll
      for (int mt = 0; mt < 2; ++mt)
#pragma unroll
        for (int nt = 0; nt < 2; ++nt)
          acc[mt * 2 + nt] = __builtin_amdgcn_wmma_f32_16x16x32_bf16(
              false, a[mt].v, false, b[nt].v, (short)0, acc[mt * 2 + nt], false, false);

      __syncthreads();   // everyone done reading buf (g&1) before refilling it

      const int gn = g + 2;                          // stream tile g+2 into buf (g&1)
      if (gn < NTILES) {
        const bf16_t* src = wbf + (size_t)(gn >> 5) * DIM * DIM + (size_t)(gn & 31) * BK;
        unsigned char* dst = lw + (gn & 1) * (BN * WSTRIDE);
#pragma unroll
        for (int c = tid; c < 512; c += 256) {
          int row = c >> 2, pos = c & 3;
          async_copy16(dst + row * WSTRIDE + pos * 16,
                       (const char*)(src + (size_t)(n0 + row) * DIM) + pos * 16);
        }
      }
    }

    // epilogue: bias + exact GELU + gated accumulation
#pragma unroll
    for (int nt = 0; nt < 2; ++nt) {
      int col = n0 + wn * 32 + nt * 16 + lm;
      float bias = eb[e * DIM + col];
#pragma unroll
      for (int mt = 0; mt < 2; ++mt) {
        int t = mt * 2 + nt;
#pragma unroll
        for (int r = 0; r < 8; ++r) {
          float y = acc[t][r] + bias;
          float gl = 0.5f * y * (1.0f + erff(y * 0.70710678118654752f));
          int row = wm * 32 + mt * 16 + r + 8 * half;
          fa[t][r] += lg[e * BM + row] * gl;
        }
      }
    }
  }

  // store: each C VGPR covers 2 contiguous 64B row segments
#pragma unroll
  for (int mt = 0; mt < 2; ++mt)
#pragma unroll
    for (int nt = 0; nt < 2; ++nt) {
      int t = mt * 2 + nt;
      int col = n0 + wn * 32 + nt * 16 + lm;
#pragma unroll
      for (int r = 0; r < 8; ++r) {
        int row = m0 + wm * 32 + mt * 16 + r + 8 * half;
        out[(size_t)row * DIM + col] = fa[t][r];
      }
    }
}

// ------------------------------- host launcher ----------------------------------
extern "C" void kernel_launch(void* const* d_in, const int* in_sizes, int n_in,
                              void* d_out, int out_size, void* d_ws, size_t ws_size,
                              hipStream_t stream) {
  const float* x  = (const float*)d_in[0];   // [N, D]
  const float* rw = (const float*)d_in[1];   // [E, D]
  const float* rb = (const float*)d_in[2];   // [E]
  const float* ew = (const float*)d_in[3];   // [E, D, D]
  const float* eb = (const float*)d_in[4];   // [E, D]
  float* out = (float*)d_out;

  char* ws = (char*)d_ws;
  bf16_t* xbf  = (bf16_t*)ws;                                        // 32 MB
  bf16_t* wbf  = (bf16_t*)(ws + (size_t)N_TOK * DIM * 2);            // 16 MB
  float*  gate = (float*)(ws + (size_t)N_TOK * DIM * 2
                             + (size_t)NEXP * DIM * DIM * 2);        // 512 KB

  int n4x = N_TOK * DIM / 4;
  cvt_bf16_kernel<<<(n4x + 255) / 256, 256, 0, stream>>>(x, (uint2*)xbf, n4x);

  int n4w = NEXP * DIM * DIM / 4;
  cvt_bf16_kernel<<<(n4w + 255) / 256, 256, 0, stream>>>(ew, (uint2*)wbf, n4w);

  router_kernel<<<N_TOK / 256, 256, 0, stream>>>(x, rw, rb, gate);

  dim3 grid(DIM / BN, N_TOK / BM);   // (8, 256)
  moe_gemm_kernel<<<grid, 256, 0, stream>>>(xbf, wbf, eb, gate, out);
}
